// SoftMoELayerWrapperMET_2972117369070
// MI455X (gfx1250) — matrix operand order repaired and checked by reading, not compiled, not run on hardware
//
#include <hip/hip_runtime.h>
#include <hip/hip_bf16.h>
#include <math.h>

typedef _Float16 v16h __attribute__((ext_vector_type(16)));
typedef _Float16 v8h  __attribute__((ext_vector_type(8)));
typedef float    v8f  __attribute__((ext_vector_type(8)));

#define BB 8
#define MMm 4096
#define DDd 1024
#define NE 8
#define PPp 3
#define SS 24            // N*P slots
#define RR 32768         // B*M token rows
#define OUT_CH 33554432  // B*M*D
#define LOSS_IDX (3 * OUT_CH)

// ---------------- reductions (wave32) ----------------
__device__ inline float waveSum(float v) {
#pragma unroll
  for (int o = 16; o > 0; o >>= 1) v += __shfl_xor(v, o, 32);
  return v;
}

__device__ inline float blockSum(float v, float* sm, int nwaves) {
  int tid = threadIdx.x, lane = tid & 31, wid = tid >> 5;
  v = waveSum(v);
  if (lane == 0) sm[wid] = v;
  __syncthreads();
  float r;
  if (wid == 0) {
    r = (lane < nwaves) ? sm[lane] : 0.f;
    r = waveSum(r);
    if (lane == 0) sm[0] = r;
  }
  __syncthreads();
  r = sm[0];
  __syncthreads();  // safe LDS reuse
  return r;
}

// ---------------- init: zero xs staging, padded A tiles, Z, loss ----------------
__global__ void k_init(float* xs, unsigned int* xsh_w, float* Z, float* loss) {
  int idx = blockIdx.x * 256 + threadIdx.x;
  if (idx < BB * SS * DDd) xs[idx] = 0.f;   // 196608 f32 accumulation buffer
  if (idx < 131072) xsh_w[idx] = 0u;        // xsh: 8 experts * 32 rows * 1024 halfs
  if (idx < BB * SS) Z[idx] = 0.f;
  if (idx == 0) *loss = 0.f;
}

// ---------------- L2-normalize tokens; emit f32 + f16 ----------------
__global__ void k_norm_x(const float* __restrict__ x, float* __restrict__ xn,
                         _Float16* __restrict__ xh) {
  __shared__ float sm[8];
  int row = blockIdx.x, tid = threadIdx.x;
  const float4* xr = (const float4*)(x + (size_t)row * DDd);
  float4 v = xr[tid];
  float ss = v.x * v.x + v.y * v.y + v.z * v.z + v.w * v.w;
  float tot = blockSum(ss, sm, 8);
  float inv = 1.f / fmaxf(sqrtf(tot), 1e-12f);
  float4 o = {v.x * inv, v.y * inv, v.z * inv, v.w * inv};
  ((float4*)(xn + (size_t)row * DDd))[tid] = o;
  _Float16* hp = xh + (size_t)row * DDd + tid * 4;
  hp[0] = (_Float16)o.x; hp[1] = (_Float16)o.y;
  hp[2] = (_Float16)o.z; hp[3] = (_Float16)o.w;
}

// ---------------- normalize phi columns -> slot-major f16, padded to 32 ----------------
__global__ void k_norm_phi(const float* __restrict__ phi, const float* __restrict__ scale,
                           _Float16* __restrict__ ph) {
  __shared__ float sm[8];
  int slot = blockIdx.x, tid = threadIdx.x;
  if (slot >= SS) {  // zero pad columns 24..31
#pragma unroll
    for (int j = 0; j < 4; ++j) ph[slot * DDd + tid + j * 256] = (_Float16)0.f;
    return;
  }
  float ss = 0.f;
#pragma unroll
  for (int j = 0; j < 4; ++j) {
    float v = phi[(tid + j * 256) * SS + slot];
    ss += v * v;
  }
  float tot = blockSum(ss, sm, 8);
  float inv = scale[0] / fmaxf(sqrtf(tot), 1e-12f);
#pragma unroll
  for (int j = 0; j < 4; ++j) {
    int d = tid + j * 256;
    ph[slot * DDd + d] = (_Float16)(phi[d * SS + slot] * inv);
  }
}

// ---------------- logits GEMM: [32768 x 1024] x [1024 x 32] via WMMA f16 ----------------
// One wave per 16-row tile computes BOTH 16-col tiles, sharing the A loads.
// A 16x32 f16 layout: lane L (m=L&15): halves 0..7 = K (k0 + 8*hs)+0..7,
//                     halves 8..15 = K (k0+16+8*hs)+0..7   (hs = L>>4)
// B 32x16 f16 layout: lane L (n=L&15): same K order as A
__global__ void k_logits(const _Float16* __restrict__ xh, const _Float16* __restrict__ ph,
                         float* __restrict__ lg) {
  int lane = threadIdx.x;
  int rt = blockIdx.x;
  int hs = lane >> 4, ml = lane & 15;
  const _Float16* ap  = xh + (size_t)(rt * 16 + ml) * DDd;
  const _Float16* bp0 = ph + (size_t)ml * DDd;
  const _Float16* bp1 = ph + (size_t)(16 + ml) * DDd;
  v8f acc0 = {}, acc1 = {};
  for (int k0 = 0; k0 < DDd; k0 += 32) {
    union { v16h v; v8h h[2]; } a;
    a.h[0] = *(const v8h*)(ap + k0 + 8 * hs);
    a.h[1] = *(const v8h*)(ap + k0 + 16 + 8 * hs);
    union { v16h v; v8h h[2]; } u0, u1;
    u0.h[0] = *(const v8h*)(bp0 + k0 + 8 * hs);
    u0.h[1] = *(const v8h*)(bp0 + k0 + 16 + 8 * hs);
    u1.h[0] = *(const v8h*)(bp1 + k0 + 8 * hs);
    u1.h[1] = *(const v8h*)(bp1 + k0 + 16 + 8 * hs);
    v16h b0 = u0.v;
    v16h b1 = u1.v;
    acc0 = __builtin_amdgcn_wmma_f32_16x16x32_f16(false, a.v, false, b0, (short)0, acc0,
                                                  false, false);
    acc1 = __builtin_amdgcn_wmma_f32_16x16x32_f16(false, a.v, false, b1, (short)0, acc1,
                                                  false, false);
  }
#pragma unroll
  for (int r = 0; r < 8; ++r) {
    int row = rt * 16 + r + 8 * hs;
    lg[(size_t)row * SS + ml] = acc0[r];          // cols 0..15 always valid
    if (ml < 8) lg[(size_t)row * SS + 16 + ml] = acc1[r];  // cols 16..23
  }
}

// ---------------- per-(b,slot) sum of exp(logits) over M ----------------
__global__ void k_zred(const float* __restrict__ lg, float* __restrict__ Z) {
  __shared__ float sm[8];
  int row = blockIdx.x * 256 + threadIdx.x;
  int b = row >> 12;
  for (int s = 0; s < SS; ++s) {
    float v = expf(lg[(size_t)row * SS + s]);
    float t = blockSum(v, sm, 8);
    if (threadIdx.x == 0) atomicAdd(&Z[b * SS + s], t);
  }
}

// ---------------- row pass: c, d, MI-loss contribution ----------------
__global__ void k_pass2(const float* __restrict__ lg, const float* __restrict__ Z,
                        float* __restrict__ dbuf, float* __restrict__ cbuf,
                        float* __restrict__ loss) {
  __shared__ float sm[8];
  int row = blockIdx.x * 256 + threadIdx.x;
  int b = row >> 12;
  float z[SS];
  float S = 0.f;
#pragma unroll
  for (int s = 0; s < SS; ++s) { z[s] = Z[b * SS + s]; S += z[s]; }
  float T[PPp] = {0.f, 0.f, 0.f};
#pragma unroll
  for (int n = 0; n < NE; ++n)
#pragma unroll
    for (int p = 0; p < PPp; ++p) T[p] += z[n * PPp + p];
  float e[SS], rowsum = 0.f;
#pragma unroll
  for (int s = 0; s < SS; ++s) { e[s] = expf(lg[(size_t)row * SS + s]); rowsum += e[s]; }
#pragma unroll
  for (int s = 0; s < SS; ++s) {
    cbuf[(size_t)row * SS + s] = e[s] / rowsum;  // softmax over slots
    dbuf[(size_t)row * SS + s] = e[s] / z[s];    // softmax over M
  }
  float lp = 0.f;
  float pm = rowsum / S;
#pragma unroll
  for (int p = 0; p < PPp; ++p) {
    float pmt = 0.f;
#pragma unroll
    for (int n = 0; n < NE; ++n) pmt += e[n * PPp + p];
    pmt /= S;
    float pt = T[p] / S;
    lp += pmt * logf(pmt / (pm * pt) + 1e-10f);
  }
  float tot = blockSum(lp, sm, 8);
  if (threadIdx.x == 0) atomicAdd(loss, -tot);
}

// ---------------- xs accumulation: each xn element read ONCE, reused for all 24 slots ----
// grid (ddchunk=4, b=8, mchunk=8); 512 rows of m per block; atomicAdd partials into xs.
__global__ void k_xs_acc(const float* __restrict__ xn, const float* __restrict__ dbuf,
                         float* __restrict__ xs) {
  __shared__ float dsh[64][SS + 1];  // +1 pad vs bank conflicts
  int tid = threadIdx.x;
  int b = blockIdx.y;
  int dd = blockIdx.x * 256 + tid;
  int mbase = blockIdx.z * 512;
  const float* xb = xn + (size_t)b * MMm * DDd;
  const float* db = dbuf + (size_t)b * MMm * SS;
  float acc[SS];
#pragma unroll
  for (int s = 0; s < SS; ++s) acc[s] = 0.f;
  for (int m0 = 0; m0 < 512; m0 += 64) {
    // prefetch next xn chunk (global_prefetch_b8)
    if (m0 + 64 < 512)
      __builtin_prefetch(&xb[(size_t)(mbase + m0 + 64) * DDd + dd], 0, 1);
    for (int i = tid; i < 64 * SS; i += 256) {
      int r = i / SS, s = i % SS;
      dsh[r][s] = db[(size_t)(mbase + m0 + r) * SS + s];
    }
    __syncthreads();
#pragma unroll 4
    for (int mm = 0; mm < 64; ++mm) {
      float xv = xb[(size_t)(mbase + m0 + mm) * DDd + dd];
#pragma unroll
      for (int s = 0; s < SS; ++s) acc[s] += dsh[mm][s] * xv;
    }
    __syncthreads();
  }
#pragma unroll
  for (int s = 0; s < SS; ++s)
    atomicAdd(&xs[(size_t)(b * SS + s) * DDd + dd], acc[s]);
}

// ---------------- xs epilogue: + task emb, emit padded f16 A tiles ----------------
__global__ void k_xs_fin(const float* __restrict__ xs, const float* __restrict__ te1,
                         const float* __restrict__ te2, const float* __restrict__ te3,
                         _Float16* __restrict__ xsh) {
  int bs = blockIdx.y;
  int b = bs / SS, slot = bs % SS;
  int n = slot / PPp, p = slot % PPp;
  int dd = blockIdx.x * 256 + threadIdx.x;
  const float* te = (p == 0) ? te1 : ((p == 1) ? te2 : te3);
  float v = xs[(size_t)(b * SS + slot) * DDd + dd] + te[dd];
  xsh[(size_t)(n * 32 + b * PPp + p) * DDd + dd] = (_Float16)v;  // rows 24..31 stay 0
}

// ---------------- expert_w [n][d][f] -> f16 [n][f][d] (K-contiguous B) ----------------
__global__ void k_wt(const float* __restrict__ ew, _Float16* __restrict__ wt) {
  int idx = blockIdx.x * 256 + threadIdx.x;  // over 8*1024*1024
  int n = idx >> 20;
  int rem = idx & 1048575;
  int f = rem >> 10, d = rem & 1023;
  wt[idx] = (_Float16)ew[((size_t)n << 20) + ((size_t)d << 10) + f];
}

// ---------------- per-expert GEMM [32 x 1024] x [1024 x 1024] via WMMA, +bias ---------
// One wave per 16-col tile computes BOTH 16-row tiles, sharing the B (weight) loads.
__global__ void k_expert(const _Float16* __restrict__ xsh, const _Float16* __restrict__ wt,
                         const float* __restrict__ eb, float* __restrict__ ys) {
  int lane = threadIdx.x;
  int n = blockIdx.x, ct = blockIdx.y;
  int hs = lane >> 4, ml = lane & 15;
  const _Float16* ap0 = xsh + (size_t)(n * 32 + ml) * DDd;
  const _Float16* ap1 = ap0 + (size_t)16 * DDd;
  const _Float16* bp  = wt + ((size_t)n << 20) + (size_t)(ct * 16 + ml) * DDd;
  v8f acc0 = {}, acc1 = {};
  for (int k0 = 0; k0 < DDd; k0 += 32) {
    union { v16h v; v8h h[2]; } a0, a1;
    a0.h[0] = *(const v8h*)(ap0 + k0 + 8 * hs);
    a0.h[1] = *(const v8h*)(ap0 + k0 + 16 + 8 * hs);
    a1.h[0] = *(const v8h*)(ap1 + k0 + 8 * hs);
    a1.h[1] = *(const v8h*)(ap1 + k0 + 16 + 8 * hs);
    union { v16h v; v8h h[2]; } ub;
    ub.h[0] = *(const v8h*)(bp + k0 + 8 * hs);
    ub.h[1] = *(const v8h*)(bp + k0 + 16 + 8 * hs);
    v16h b = ub.v;
    acc0 = __builtin_amdgcn_wmma_f32_16x16x32_f16(false, a0.v, false, b, (short)0, acc0,
                                                  false, false);
    acc1 = __builtin_amdgcn_wmma_f32_16x16x32_f16(false, a1.v, false, b, (short)0, acc1,
                                                  false, false);
  }
  int col = ct * 16 + ml;
  float bias = eb[n * DDd + col];
#pragma unroll
  for (int r = 0; r < 8; ++r) {
    int rowA = r + 8 * hs;  // 0..15, always valid (< 24)
    int bi = rowA / PPp, p = rowA % PPp;
    ys[(size_t)(bi * SS + n * PPp + p) * DDd + col] = acc0[r] + bias;
  }
#pragma unroll
  for (int r = 0; r < 8; ++r) {
    int rowA = 16 + r + 8 * hs;  // 16..31, valid if < 24
    if (rowA < SS) {
      int bi = rowA / PPp, p = rowA % PPp;
      ys[(size_t)(bi * SS + n * PPp + p) * DDd + col] = acc1[r] + bias;
    }
  }
}

// ---------------- final combine: y[b,m,p,:] = sum_n c[b,m,n,p] * ys[b,n,p,:] ----------
__global__ void k_out(const float* __restrict__ cbuf, const float* __restrict__ ys,
                      float* __restrict__ out) {
  __shared__ float cL[SS];
  int row = blockIdx.x;
  int b = row >> 12;
  int dd = blockIdx.y * 256 + threadIdx.x;
  if (threadIdx.x < SS) cL[threadIdx.x] = cbuf[(size_t)row * SS + threadIdx.x];
  __syncthreads();
  const float* yb = ys + (size_t)b * SS * DDd;
#pragma unroll
  for (int p = 0; p < PPp; ++p) {
    float a = 0.f;
#pragma unroll
    for (int n = 0; n < NE; ++n) a += cL[n * PPp + p] * yb[(size_t)(n * PPp + p) * DDd + dd];
    out[(size_t)p * OUT_CH + (size_t)row * DDd + dd] = a;
  }
}

extern "C" void kernel_launch(void* const* d_in, const int* in_sizes, int n_in,
                              void* d_out, int out_size, void* d_ws, size_t ws_size,
                              hipStream_t stream) {
  (void)in_sizes; (void)n_in; (void)out_size; (void)ws_size;
  const float* x   = (const float*)d_in[0];
  const float* phi = (const float*)d_in[1];
  const float* scl = (const float*)d_in[2];
  const float* te1 = (const float*)d_in[3];
  const float* te2 = (const float*)d_in[4];
  const float* te3 = (const float*)d_in[5];
  const float* ew  = (const float*)d_in[6];
  const float* eb  = (const float*)d_in[7];
  float* out = (float*)d_out;

  char* w = (char*)d_ws;
  float*    xn   = (float*)(w + 0);                  // 128 MB
  _Float16* xh   = (_Float16*)(w + 134217728);       // 64 MB
  _Float16* ph   = (_Float16*)(w + 201326592);       // 64 KB (32x1024 f16, slot-major)
  float*    lg   = (float*)(w + 201392128);          // logits [32768][24]
  float*    dbuf = (float*)(w + 204537856);          // dispatch [32768][24]
  float*    cbuf = (float*)(w + 207683584);          // combine  [32768][24]
  float*    Z    = (float*)(w + 210829312);          // [8][24]
  float*    xs   = (float*)(w + 210830336);          // [8][24][1024] f32 staging
  _Float16* xsh  = (_Float16*)(w + 211616768);       // [8][32][1024] f16 (padded A)
  _Float16* wt   = (_Float16*)(w + 212141056);       // [8][1024][1024] f16 (f-major)
  float*    ys   = (float*)(w + 228918272);          // [8][24][1024] f32

  k_init<<<768, 256, 0, stream>>>(xs, (unsigned int*)xsh, Z, out + LOSS_IDX);
  k_norm_x<<<RR, 256, 0, stream>>>(x, xn, xh);
  k_norm_phi<<<32, 256, 0, stream>>>(phi, scl, ph);
  k_logits<<<RR / 16, 32, 0, stream>>>(xh, ph, lg);
  k_zred<<<RR / 256, 256, 0, stream>>>(lg, Z);
  k_pass2<<<RR / 256, 256, 0, stream>>>(lg, Z, dbuf, cbuf, out + LOSS_IDX);
  k_xs_acc<<<dim3(4, BB, 8), 256, 0, stream>>>(xn, dbuf, xs);
  k_xs_fin<<<dim3(4, BB * SS), 256, 0, stream>>>(xs, te1, te2, te3, xsh);
  k_wt<<<32768, 256, 0, stream>>>(ew, wt);
  k_expert<<<dim3(NE, DDd / 16), 32, 0, stream>>>(xsh, wt, eb, ys);
  k_out<<<dim3(RR, 4), 256, 0, stream>>>(cbuf, ys, out);
}
